// AttnBlock_67396626808938
// MI455X (gfx1250) — matrix operand order repaired and checked by reading, not compiled, not run on hardware
//
#include <hip/hip_runtime.h>
#include <hip/hip_bf16.h>

// ---------------------------------------------------------------------------
// AttnBlock for MI455X (gfx1250, wave32, WMMA bf16 16x16x32 w/ f32 accum)
// C=512, H=W=64, S=4096, GROUPS=32
// All GEMM operands laid out K-contiguous so every fragment load is 2x b128.
// attnv_gemm stages its shared B tile through LDS with async loads (ASYNCcnt).
// ---------------------------------------------------------------------------

#define C_DIM 512
#define S_DIM 4096
#define GROUPS 32
#define CPG (C_DIM / GROUPS)   // 16 channels per group

typedef __attribute__((ext_vector_type(16))) __bf16 v16bf;
typedef __attribute__((ext_vector_type(8)))  float  v8f;
typedef __attribute__((ext_vector_type(4)))  unsigned int v4u;
typedef __attribute__((ext_vector_type(4)))  int v4i;

// Address-space qualified element types for the async-LDS builtin
typedef __attribute__((address_space(1))) v4i v4i_glob;
typedef __attribute__((address_space(3))) v4i v4i_lds;

union Frag { v16bf v; unsigned short u[16]; unsigned int d[8]; v4u q[2]; };

#if __has_builtin(__builtin_amdgcn_global_load_async_to_lds_b128)
#define HAVE_ASYNC_LDS 1
#else
#define HAVE_ASYNC_LDS 0
#endif

__device__ __forceinline__ unsigned short f2bf(float f) {
    unsigned int u = __float_as_uint(f);
    u += 0x7FFFu + ((u >> 16) & 1u);   // round-to-nearest-even
    return (unsigned short)(u >> 16);
}

// Starting K (within a 32-wide K block) of the bf16 pair held in dword v of the
// A fragment: lanes 0-15 hold K {0..15},
// lanes 16-31 hold K {16..31}; each dword = 2 consecutive K.
__device__ __forceinline__ int a_kpair(int lane, int v) {
    int k = 2 * v;
    return (lane < 16) ? k : (k + 16);
}

// A fragment: row `arow` (points at K-block start), K contiguous in memory.
__device__ __forceinline__ void load_afrag(const unsigned short* __restrict__ arow,
                                           int lane, Frag& f) {
#pragma unroll
    for (int v = 0; v < 8; ++v)
        f.d[v] = *(const unsigned int*)(arow + a_kpair(lane, v));
}

// B fragment (32x16): lane holds column n = lane&15, K = e + (lane<16 ? 0:16).
// `brow` points at element K=(lane<16 ? 0:16) of this lane's column; K contiguous in memory.
__device__ __forceinline__ void load_bfrag(const unsigned short* __restrict__ brow,
                                           Frag& f) {
#pragma unroll
    for (int v = 0; v < 8; ++v)
        f.d[v] = *(const unsigned int*)(brow + 2 * v);
}

__device__ __forceinline__ v8f wmma_bf16(const Frag& a, const Frag& b, v8f c) {
    return __builtin_amdgcn_wmma_f32_16x16x32_bf16(
        false, a.v, false, b.v, (short)0, c, false, false);
}

// 16-byte global -> LDS copy: async on CDNA5 toolchains that expose the builtin
// (lowers to global_load_async_to_lds_b128, ASYNCcnt), else staged via VGPRs.
__device__ __forceinline__ void cp16_to_lds(const unsigned short* g, unsigned char* l) {
#if HAVE_ASYNC_LDS
    __builtin_amdgcn_global_load_async_to_lds_b128(
        (v4i_glob*)g, (v4i_lds*)l, 0, 0);
#else
    *(v4u*)l = *(const v4u*)g;
#endif
}

__device__ __forceinline__ void async_wait_le2() {
#if HAVE_ASYNC_LDS
#if __has_builtin(__builtin_amdgcn_s_wait_asynccnt)
    __builtin_amdgcn_s_wait_asynccnt(2);
#else
    asm volatile("s_wait_asynccnt 0x2" ::: "memory");
#endif
#endif
}

__device__ __forceinline__ void async_wait_le0() {
#if HAVE_ASYNC_LDS
#if __has_builtin(__builtin_amdgcn_s_wait_asynccnt)
    __builtin_amdgcn_s_wait_asynccnt(0);
#else
    asm volatile("s_wait_asynccnt 0x0" ::: "memory");
#endif
#endif
}

// ---------------------------------------------------------------------------
// f32 -> bf16 convert (weights)
// ---------------------------------------------------------------------------
__global__ void cvt_bf16(const float* __restrict__ src, unsigned short* __restrict__ dst, int n) {
    int i = blockIdx.x * blockDim.x + threadIdx.x;
    if (i < n) dst[i] = f2bf(src[i]);
}

// ---------------------------------------------------------------------------
// GroupNorm stats: one block per group, reduce 16*4096 elements
// ---------------------------------------------------------------------------
__global__ void gn_stats(const float* __restrict__ x, float* __restrict__ stats) {
    const int g = blockIdx.x;
    const float* base = x + (size_t)g * CPG * S_DIM;
    float s = 0.f, s2 = 0.f;
    for (int i = threadIdx.x; i < CPG * S_DIM; i += 256) {
        float v = base[i];
        s += v; s2 += v * v;
    }
    __shared__ float sh[256], sh2[256];
    sh[threadIdx.x] = s; sh2[threadIdx.x] = s2;
    __syncthreads();
    for (int o = 128; o > 0; o >>= 1) {
        if (threadIdx.x < (unsigned)o) {
            sh[threadIdx.x]  += sh[threadIdx.x + o];
            sh2[threadIdx.x] += sh2[threadIdx.x + o];
        }
        __syncthreads();
    }
    if (threadIdx.x == 0) {
        const float inv_n = 1.0f / (float)(CPG * S_DIM);
        float mean = sh[0] * inv_n;
        float var  = sh2[0] * inv_n - mean * mean;
        stats[g] = mean;
        stats[GROUPS + g] = rsqrtf(var + 1e-6f);
    }
}

// ---------------------------------------------------------------------------
// GroupNorm apply; emit bf16 normalized activations TRANSPOSED: hn_t[S][C]
// ---------------------------------------------------------------------------
__global__ void gn_apply(const float* __restrict__ x, const float* __restrict__ scale,
                         const float* __restrict__ offset, const float* __restrict__ stats,
                         unsigned short* __restrict__ hn_t) {
    size_t idx = (size_t)blockIdx.x * blockDim.x + threadIdx.x;   // < 512*4096
    int c = (int)(idx >> 12);          // x is [C][S]
    int s = (int)(idx & (S_DIM - 1));
    int g = c >> 4;
    float m = stats[g], r = stats[GROUPS + g];
    float v = (x[idx] - m) * r * scale[c] + offset[c];
    hn_t[(size_t)s * C_DIM + c] = f2bf(v);
}

// ---------------------------------------------------------------------------
// Projection GEMM (q/k/v): acc[o,s] = sum_i W[o,i]*Hn[i,s] + bias[o]
//   W    : [512,512]  bf16 row-major       (A, K contiguous)
//   Hn_t : [4096,512] bf16 row-major       (B, K contiguous)
//   TRANS=1 -> store O_t[s][o] (for q,k);  TRANS=0 -> store O[o][s] (for v)
// Wave: 32x64 strip (8 WMMA / K-step). Block 128 thr = 4 waves -> 128x64 tile.
// ---------------------------------------------------------------------------
template <bool TRANS>
__global__ __launch_bounds__(128) void proj_gemm(
        const unsigned short* __restrict__ Wb, const unsigned short* __restrict__ Hn_t,
        const float* __restrict__ bias, unsigned short* __restrict__ Ob) {
    const int lane = threadIdx.x & 31;
    const int wave = threadIdx.x >> 5;
    const int mbase = blockIdx.y * 128 + wave * 32;
    const int nbase = blockIdx.x * 64;
    const int mrow  = mbase + (lane & 15);
    const int kb_b  = (lane < 16) ? 0 : 16;
    const int ncol  = lane & 15;

    v8f acc[2][4] = {};
    for (int k0 = 0; k0 < C_DIM; k0 += 32) {
        Frag a0, a1;
        load_afrag(Wb + (size_t)mrow * C_DIM + k0, lane, a0);
        load_afrag(Wb + (size_t)(mrow + 16) * C_DIM + k0, lane, a1);
#pragma unroll
        for (int j = 0; j < 4; ++j) {
            Frag b;
            load_bfrag(Hn_t + (size_t)(nbase + j * 16 + ncol) * C_DIM + k0 + kb_b, b);
            acc[0][j] = wmma_bf16(a0, b, acc[0][j]);
            acc[1][j] = wmma_bf16(a1, b, acc[1][j]);
        }
    }
#pragma unroll
    for (int i = 0; i < 2; ++i) {
#pragma unroll
        for (int j = 0; j < 4; ++j) {
#pragma unroll
            for (int e = 0; e < 8; ++e) {
                int m = mbase + i * 16 + e + ((lane >= 16) ? 8 : 0);
                int n = nbase + j * 16 + ncol;
                unsigned short val = f2bf(acc[i][j][e] + bias[m]);
                if (TRANS) Ob[(size_t)n * C_DIM + m] = val;
                else       Ob[(size_t)m * S_DIM + n] = val;
            }
        }
    }
}

// ---------------------------------------------------------------------------
// Logits GEMM: P[t,s] = alpha * sum_f q[f,t]*k[f,s]
//   Qt, Kt : [4096,512] bf16 (both K contiguous).  P fp32 (64 MB, L2-resident)
// ---------------------------------------------------------------------------
__global__ __launch_bounds__(128) void logits_gemm(
        const unsigned short* __restrict__ Qt, const unsigned short* __restrict__ Kt,
        float alpha, float* __restrict__ P) {
    const int lane = threadIdx.x & 31;
    const int wave = threadIdx.x >> 5;
    const int tbase = blockIdx.y * 128 + wave * 32;
    const int sbase = blockIdx.x * 64;
    const int trow  = tbase + (lane & 15);
    const int kb_b  = (lane < 16) ? 0 : 16;
    const int ncol  = lane & 15;

    v8f acc[2][4] = {};
    for (int k0 = 0; k0 < C_DIM; k0 += 32) {
        Frag a0, a1;
        load_afrag(Qt + (size_t)trow * C_DIM + k0, lane, a0);
        load_afrag(Qt + (size_t)(trow + 16) * C_DIM + k0, lane, a1);
#pragma unroll
        for (int j = 0; j < 4; ++j) {
            Frag b;
            load_bfrag(Kt + (size_t)(sbase + j * 16 + ncol) * C_DIM + k0 + kb_b, b);
            acc[0][j] = wmma_bf16(a0, b, acc[0][j]);
            acc[1][j] = wmma_bf16(a1, b, acc[1][j]);
        }
    }
#pragma unroll
    for (int i = 0; i < 2; ++i) {
#pragma unroll
        for (int j = 0; j < 4; ++j) {
#pragma unroll
            for (int e = 0; e < 8; ++e) {
                int t = tbase + i * 16 + e + ((lane >= 16) ? 8 : 0);
                int s = sbase + j * 16 + ncol;
                P[(size_t)t * S_DIM + s] = alpha * acc[i][j][e];
            }
        }
    }
}

// ---------------------------------------------------------------------------
// Row softmax (fp32 in, normalized bf16 out), one block per row t
// ---------------------------------------------------------------------------
__global__ void softmax_row(const float* __restrict__ P, unsigned short* __restrict__ Wt) {
    const int t = blockIdx.x;
    const float* row = P + (size_t)t * S_DIM;
    __shared__ float sh[256];

    float mx = -3.4e38f;
    for (int i = threadIdx.x; i < S_DIM; i += 256) mx = fmaxf(mx, row[i]);
    sh[threadIdx.x] = mx; __syncthreads();
    for (int o = 128; o > 0; o >>= 1) {
        if (threadIdx.x < (unsigned)o) sh[threadIdx.x] = fmaxf(sh[threadIdx.x], sh[threadIdx.x + o]);
        __syncthreads();
    }
    mx = sh[0]; __syncthreads();

    float s = 0.f;
    for (int i = threadIdx.x; i < S_DIM; i += 256) s += __expf(row[i] - mx);
    sh[threadIdx.x] = s; __syncthreads();
    for (int o = 128; o > 0; o >>= 1) {
        if (threadIdx.x < (unsigned)o) sh[threadIdx.x] += sh[threadIdx.x + o];
        __syncthreads();
    }
    const float inv = 1.0f / sh[0];

    unsigned short* orow = Wt + (size_t)t * S_DIM;
    for (int i = threadIdx.x; i < S_DIM; i += 256)
        orow[i] = f2bf(__expf(row[i] - mx) * inv);
}

// ---------------------------------------------------------------------------
// Attention x V GEMM: h[f,t] = sum_s V[f,s] * Wt[t,s]   (K = 4096)
//   V  : [512,4096] bf16 (A, K contiguous, direct global)
//   Wt : [4096,4096] bf16 (B tile shared by all 4 waves -> staged in LDS via
//        async global->LDS copies, double-buffered on ASYNCcnt)
//   Store TRANSPOSED: Ht[t][f]  (so final proj B is K contiguous)
// LDS stage: 64 rows x 64B (K=32), padded to 80B/row for bank-conflict-free
// b128 reads (n*20 dwords mod 64 distinct for n=0..15).
// ---------------------------------------------------------------------------
#define STAGE_ROW_B 80
#define STAGE_BYTES (64 * STAGE_ROW_B)

__global__ __launch_bounds__(128) void attnv_gemm(
        const unsigned short* __restrict__ V, const unsigned short* __restrict__ Wt,
        unsigned short* __restrict__ Ht) {
    const int lane = threadIdx.x & 31;
    const int wave = threadIdx.x >> 5;
    const int fbase = blockIdx.y * 128 + wave * 32;
    const int tbase = blockIdx.x * 64;
    const int frow  = fbase + (lane & 15);
    const int kb_b  = (lane < 16) ? 0 : 16;
    const int ncol  = lane & 15;

    __shared__ __align__(16) unsigned char smem[2 * STAGE_BYTES];

    // Stage the 64x32 B tile at K offset k0 into LDS buffer s.
    // 4 KB of payload = 256 x 16B chunks; 128 threads x 2 chunks each.
    auto issue_stage = [&](int s, int k0) {
        unsigned char* st = smem + s * STAGE_BYTES;
#pragma unroll
        for (int i = 0; i < 2; ++i) {
            int q = (int)threadIdx.x * 2 + i;
            int r = q >> 2;          // tile row 0..63
            int c = q & 3;           // 16B chunk within row
            cp16_to_lds(Wt + (size_t)(tbase + r) * S_DIM + k0 + c * 8,
                        st + r * STAGE_ROW_B + c * 16);
        }
    };

    v8f acc[2][4] = {};

    auto compute = [&](int s, int k0) {
        const unsigned char* st = smem + s * STAGE_BYTES;
        Frag a0, a1;
        load_afrag(V + (size_t)frow * S_DIM + k0, lane, a0);
        load_afrag(V + (size_t)(frow + 16) * S_DIM + k0, lane, a1);
#pragma unroll
        for (int j = 0; j < 4; ++j) {
            Frag b;
            const unsigned char* lrow = st + (j * 16 + ncol) * STAGE_ROW_B + kb_b * 2;
            b.q[0] = *(const v4u*)(lrow);
            b.q[1] = *(const v4u*)(lrow + 16);
            acc[0][j] = wmma_bf16(a0, b, acc[0][j]);
            acc[1][j] = wmma_bf16(a1, b, acc[1][j]);
        }
    };

    issue_stage(0, 0);
    int buf = 0;
    for (int k0 = 0; k0 < S_DIM - 32; k0 += 32) {
        issue_stage(buf ^ 1, k0 + 32);   // prefetch next tile (2 async ops/thread)
        async_wait_le2();                // current tile's 2 ops have completed
        __syncthreads();
        compute(buf, k0);
        __syncthreads();                 // everyone done reading before reuse
        buf ^= 1;
    }
    async_wait_le0();
    __syncthreads();
    compute(buf, S_DIM - 32);

#pragma unroll
    for (int i = 0; i < 2; ++i) {
#pragma unroll
        for (int j = 0; j < 4; ++j) {
#pragma unroll
            for (int e = 0; e < 8; ++e) {
                int f = fbase + i * 16 + e + ((lane >= 16) ? 8 : 0);
                int t = tbase + j * 16 + ncol;
                Ht[(size_t)t * C_DIM + f] = f2bf(acc[i][j][e]);
            }
        }
    }
}

// ---------------------------------------------------------------------------
// Output projection + residual: out[o,s] = x[o,s] + sum_i Wo[o,i]*H[i,s] + bo[o]
//   Wo : [512,512] bf16 (K contiguous),  Ht : [4096,512] bf16 (K contiguous)
// ---------------------------------------------------------------------------
__global__ __launch_bounds__(128) void final_gemm(
        const unsigned short* __restrict__ Wb, const unsigned short* __restrict__ Ht,
        const float* __restrict__ bias, const float* __restrict__ xres,
        float* __restrict__ out) {
    const int lane = threadIdx.x & 31;
    const int wave = threadIdx.x >> 5;
    const int mbase = blockIdx.y * 128 + wave * 32;
    const int nbase = blockIdx.x * 64;
    const int mrow  = mbase + (lane & 15);
    const int kb_b  = (lane < 16) ? 0 : 16;
    const int ncol  = lane & 15;

    v8f acc[2][4] = {};
    for (int k0 = 0; k0 < C_DIM; k0 += 32) {
        Frag a0, a1;
        load_afrag(Wb + (size_t)mrow * C_DIM + k0, lane, a0);
        load_afrag(Wb + (size_t)(mrow + 16) * C_DIM + k0, lane, a1);
#pragma unroll
        for (int j = 0; j < 4; ++j) {
            Frag b;
            load_bfrag(Ht + (size_t)(nbase + j * 16 + ncol) * C_DIM + k0 + kb_b, b);
            acc[0][j] = wmma_bf16(a0, b, acc[0][j]);
            acc[1][j] = wmma_bf16(a1, b, acc[1][j]);
        }
    }
#pragma unroll
    for (int i = 0; i < 2; ++i) {
#pragma unroll
        for (int j = 0; j < 4; ++j) {
#pragma unroll
            for (int e = 0; e < 8; ++e) {
                int m = mbase + i * 16 + e + ((lane >= 16) ? 8 : 0);
                int n = nbase + j * 16 + ncol;
                size_t idx = (size_t)m * S_DIM + n;
                out[idx] = xres[idx] + acc[i][j][e] + bias[m];
            }
        }
    }
}

// ---------------------------------------------------------------------------
// Host-side orchestration
// ---------------------------------------------------------------------------
extern "C" void kernel_launch(void* const* d_in, const int* in_sizes, int n_in,
                              void* d_out, int out_size, void* d_ws, size_t ws_size,
                              hipStream_t stream) {
    const float* x        = (const float*)d_in[0];
    const float* gn_scale = (const float*)d_in[1];
    const float* gn_off   = (const float*)d_in[2];
    const float* wq = (const float*)d_in[3];  const float* bq = (const float*)d_in[4];
    const float* wk = (const float*)d_in[5];  const float* bk = (const float*)d_in[6];
    const float* wv = (const float*)d_in[7];  const float* bv = (const float*)d_in[8];
    const float* wo = (const float*)d_in[9];  const float* bo = (const float*)d_in[10];

    const size_t WSZ = (size_t)C_DIM * C_DIM;        // 262144 weight elems
    const size_t HSZ = (size_t)C_DIM * S_DIM;        // 2M activation elems
    const size_t SSZ = (size_t)S_DIM * S_DIM;        // 16M attention elems

    char* ws = (char*)d_ws;
    float*          stats   = (float*)ws;                                 // 64 f
    unsigned short* wbf     = (unsigned short*)(ws + 1024);               // 4*WSZ bf16
    unsigned short* wq_b    = wbf + 0 * WSZ;
    unsigned short* wk_b    = wbf + 1 * WSZ;
    unsigned short* wv_b    = wbf + 2 * WSZ;
    unsigned short* wo_b    = wbf + 3 * WSZ;
    unsigned short* hn_t    = wbf + 4 * WSZ;                              // [S][C] bf16
    unsigned short* q_t     = hn_t + HSZ;                                 // [S][C] bf16
    unsigned short* k_t     = q_t  + HSZ;                                 // [S][C] bf16
    unsigned short* v_m     = k_t  + HSZ;                                 // [C][S] bf16
    unsigned short* h_t     = v_m  + HSZ;                                 // [S][C] bf16
    float*          logits  = (float*)(h_t + HSZ);                        // [S][S] f32
    unsigned short* attw    = (unsigned short*)(logits + SSZ);            // [S][S] bf16

    // 1) weights -> bf16
    cvt_bf16<<<(int)(WSZ / 256), 256, 0, stream>>>(wq, wq_b, (int)WSZ);
    cvt_bf16<<<(int)(WSZ / 256), 256, 0, stream>>>(wk, wk_b, (int)WSZ);
    cvt_bf16<<<(int)(WSZ / 256), 256, 0, stream>>>(wv, wv_b, (int)WSZ);
    cvt_bf16<<<(int)(WSZ / 256), 256, 0, stream>>>(wo, wo_b, (int)WSZ);

    // 2) GroupNorm (emit transposed bf16 activations)
    gn_stats<<<GROUPS, 256, 0, stream>>>(x, stats);
    gn_apply<<<(int)(HSZ / 256), 256, 0, stream>>>(x, gn_scale, gn_off, stats, hn_t);

    // 3) q/k/v projections (WMMA bf16); q,k stored [S][C], v stored [C][S]
    dim3 gp(S_DIM / 64, C_DIM / 128);   // (64, 4)
    proj_gemm<true ><<<gp, 128, 0, stream>>>(wq_b, hn_t, bq, q_t);
    proj_gemm<true ><<<gp, 128, 0, stream>>>(wk_b, hn_t, bk, k_t);
    proj_gemm<false><<<gp, 128, 0, stream>>>(wv_b, hn_t, bv, v_m);

    // 4) logits = q^T k / sqrt(C)  (fp32, lives in L2)
    const float alpha = 1.0f / sqrtf((float)C_DIM);
    dim3 gl(S_DIM / 64, S_DIM / 128);   // (64, 32)
    logits_gemm<<<gl, 128, 0, stream>>>(q_t, k_t, alpha, logits);

    // 5) softmax over keys, emit bf16 weights
    softmax_row<<<S_DIM, 256, 0, stream>>>(logits, attw);

    // 6) h = V * attw^T (async-LDS staged B), stored transposed [T][C]
    dim3 ga(S_DIM / 64, C_DIM / 128);
    attnv_gemm<<<ga, 128, 0, stream>>>(v_m, attw, h_t);

    // 7) output projection + bias + residual
    final_gemm<<<gp, 128, 0, stream>>>(wo_b, h_t, bo, x, (float*)d_out);
}
